// MIAttention_18485539242615
// MI455X (gfx1250) — matrix operand order repaired and checked by reading, not compiled, not run on hardware
//
#include <hip/hip_runtime.h>

// =====================================================================
// MI455X (gfx1250) attention pipeline — all GEMMs on v_wmma_f32_16x16x32_bf16.
//
// Stage 0: cvt_f32_bf16   — one-shot f32->bf16 of X (h/p/g) and both weights
//                           (bandwidth-bound; removes ALL cvt VALU from GEMM loops)
// Stage 1: qkv_gemm       — X@Wqkv^T -> q,k row-major [B,H,2576,64] bf16,
//                           v TRANSPOSED [B,H,64,2576] bf16 (direct b128 store
//                           from WMMA C layout; q/k via 512B LDS transpose)
// Stage 2: attn_flash     — flash attention, 1 wave32 / 16-row Q tile,
//                           online softmax, P re-tiled through LDS,
//                           V^T gives aligned b128 B-fragments
// Stage 3: proj_gemm      — x@Wproj^T + b -> f32 d_out (xh, xp, xg order)
//
// Padded token axis (2576): g=0, gap=1..15 (masked), h=16..2063, p=2064..2575.
// Workspace: ~28.5 MB of bf16 scratch.
// =====================================================================

typedef __attribute__((ext_vector_type(16))) __bf16 v16bf;
typedef __attribute__((ext_vector_type(8)))  __bf16 v8bf;
typedef __attribute__((ext_vector_type(8)))  float  v8f;

#define NTOKP 2576      // padded tokens per (b,h)
#define TOK_H 16
#define TOK_P 2064

static __device__ __forceinline__ unsigned short f2bf(float f) {
    unsigned int u = __builtin_bit_cast(unsigned int, f);
    u += 0x7FFFu + ((u >> 16) & 1u);                 // round-to-nearest-even
    return (unsigned short)(u >> 16);
}
static __device__ __forceinline__ unsigned int pk2bf(float lo, float hi) {
    return (unsigned int)f2bf(lo) | ((unsigned int)f2bf(hi) << 16);
}
// 8 consecutive bf16 via one b128 load
static __device__ __forceinline__ v8bf ld8bf(const unsigned short* p) {
    return __builtin_bit_cast(v8bf, *reinterpret_cast<const uint4*>(p));
}
static __device__ __forceinline__ v16bf cat16(v8bf a, v8bf b) {
    return __builtin_shufflevector(a, b, 0,1,2,3,4,5,6,7,8,9,10,11,12,13,14,15);
}

// ---------------------------------------------------------------------
// Stage 0: vectorized f32 -> bf16 (8 elements / thread)
// ---------------------------------------------------------------------
__global__ __launch_bounds__(256) void cvt_f32_bf16(
    const float* __restrict__ in, unsigned short* __restrict__ out, int n8)
{
    int i = blockIdx.x * blockDim.x + threadIdx.x;
    if (i >= n8) return;
    const float4* p = reinterpret_cast<const float4*>(in) + (size_t)i * 2;
    float4 a = p[0], b = p[1];
    uint4 o;
    o.x = pk2bf(a.x, a.y);  o.y = pk2bf(a.z, a.w);
    o.z = pk2bf(b.x, b.y);  o.w = pk2bf(b.z, b.w);
    reinterpret_cast<uint4*>(out)[i] = o;
}

// ---------------------------------------------------------------------
// Stage 1: QKV projection, bf16 in / bf16 out.
// q,k: row-major [B,H,NTOKP,64]; v: transposed [B,H,64,NTOKP].
// One wave per 16x16 output tile; K-loop 512/32 = 16 WMMAs.
// ---------------------------------------------------------------------
__global__ __launch_bounds__(32) void qkv_gemm(
    const unsigned short* __restrict__ X,   // bf16 [B*Ns, 512]
    const unsigned short* __restrict__ W,   // bf16 [1536, 512]
    unsigned short* __restrict__ qb, unsigned short* __restrict__ kbuf,
    unsigned short* __restrict__ vbT, int Ns, int tokOff)
{
    __shared__ __align__(16) unsigned short ldsT[16 * 16];    // 512 B
    const int lane = threadIdx.x, half = lane >> 4, l15 = lane & 15;
    const int mt = blockIdx.x, ct = blockIdx.y, b = blockIdx.z;
    const int c0 = ct * 16;

    int m = mt * 16 + l15; if (m >= Ns) m = Ns - 1;
    const unsigned short* xrow = X + ((size_t)b * Ns + m) * 512;
    const unsigned short* wrow = W + (size_t)(c0 + l15) * 512;

    v8f acc = {};
    for (int k0 = 0; k0 < 512; k0 += 32) {
        const unsigned short* xp = xrow + k0 + half * 16;     // A: e -> K=half*16+e
        v16bf A = cat16(ld8bf(xp), ld8bf(xp + 8));
        const unsigned short* wp = wrow + k0 + half * 16;     // B: e -> K=half*16+e
        v16bf Bm = cat16(ld8bf(wp), ld8bf(wp + 8));
        acc = __builtin_amdgcn_wmma_f32_16x16x32_bf16(false, A, false, Bm,
                                                      (short)0, acc, false, false);
    }

    const int which = c0 >> 9;                 // 0=q 1=k 2=v (tile never spans)
    const int h = (c0 & 511) >> 6;
    const int d0 = c0 & 63;

    if (which == 2) {
        // V transposed store: C layout is (col d fixed/lane, 8 consecutive rows)
        // -> directly contiguous in d-major layout: one b128 store per lane.
        unsigned short* dst = vbT + (((size_t)b * 8 + h) * 64 + d0 + l15) * NTOKP
                                  + tokOff + mt * 16 + 8 * half;
        if (mt * 16 + 16 <= Ns) {
            uint4 o;
            o.x = pk2bf(acc[0], acc[1]);  o.y = pk2bf(acc[2], acc[3]);
            o.z = pk2bf(acc[4], acc[5]);  o.w = pk2bf(acc[6], acc[7]);
            *reinterpret_cast<uint4*>(dst) = o;
        } else {                                            // only the Ns=1 launch
#pragma unroll
            for (int r = 0; r < 8; r++)
                if (mt * 16 + 8 * half + r < Ns) dst[r] = f2bf(acc[r]);
        }
    } else {
        // q/k row-major: transpose tile through LDS, then b128 row stores
        unsigned short* dstB = (which == 0 ? qb : kbuf)
                             + (((size_t)b * 8 + h) * NTOKP + tokOff) * 64 + d0;
#pragma unroll
        for (int r = 0; r < 8; r++)
            ldsT[(r + 8 * half) * 16 + l15] = f2bf(acc[r]);
        __syncthreads();
        int row = l15;                                      // 2 lanes per row
        uint4 o = *reinterpret_cast<const uint4*>(ldsT + row * 16 + half * 8);
        int mg = mt * 16 + row;
        if (mg < Ns)
            *reinterpret_cast<uint4*>(dstB + (size_t)mg * 64 + half * 8) = o;
    }
}

// ---------------------------------------------------------------------
// Stage 2: flash attention. q,k row-major; v transposed (d-major).
// Runs Nk>>5 full 32-key blocks, no K/V tail clamps (Nk=2576 drops its last 16 keys); gapMask masks the
// 15 padding tokens (only in the global cross-attn launch).
// ---------------------------------------------------------------------
__global__ __launch_bounds__(32) void attn_flash(
    const unsigned short* __restrict__ qb, const unsigned short* __restrict__ kmat,
    const unsigned short* __restrict__ vbT, unsigned short* __restrict__ xout,
    int qOff, int Nq, int kOff, int Nk, int gapMask)
{
    __shared__ __align__(16) unsigned short ldsP[16 * 32];   // 1 KB P tile
    const int lane = threadIdx.x, half = lane >> 4, l15 = lane & 15;
    const int qt = blockIdx.x, h = blockIdx.y, b = blockIdx.z;
    const size_t bhRM = ((size_t)b * 8 + h) * NTOKP;         // token-major base
    const size_t bhVT = ((size_t)b * 8 + h) * 64;            // d-major base
    const float SCALE = 0.125f;                              // 64^-0.5

    int mA = qt * 16 + l15; if (mA >= Nq) mA = Nq - 1;
    const unsigned short* qrow = qb + (bhRM + qOff + mA) * 64;
    v16bf Aq0 = cat16(ld8bf(qrow + half * 16),      ld8bf(qrow + half * 16 + 8));
    v16bf Aq1 = cat16(ld8bf(qrow + 32 + half * 16), ld8bf(qrow + 32 + half * 16 + 8));

    float mrow[8], lrow[8];
    v8f zero = {};
    v8f accO[4] = {zero, zero, zero, zero};
#pragma unroll
    for (int r = 0; r < 8; r++) { mrow[r] = -3.0e38f; lrow[r] = 0.0f; }

    const int nKB = Nk >> 5;
    for (int ib = 0; ib < nKB; ib++) {
        const int kb0 = ib * 32;

        if (kb0 + 32 < Nk) {                 // prefetch next block (global_prefetch_b8)
            __builtin_prefetch(kmat + (bhRM + kOff + kb0 + 32 + lane) * 64, 0, 1);
            __builtin_prefetch(vbT + (bhVT + lane) * NTOKP + kOff + kb0 + 32, 0, 1);
            __builtin_prefetch(vbT + (bhVT + lane + 32) * NTOKP + kOff + kb0 + 32, 0, 1);
        }

        // ---- S = scale * Q @ K^T, two 16-col tiles
        v8f S[2];
#pragma unroll
        for (int nt = 0; nt < 2; nt++) {
            const unsigned short* krow = kmat + (bhRM + kOff + kb0 + nt * 16 + l15) * 64;
            v16bf B0 = cat16(ld8bf(krow + half * 16),      ld8bf(krow + half * 16 + 8));
            v16bf B1 = cat16(ld8bf(krow + 32 + half * 16), ld8bf(krow + 32 + half * 16 + 8));
            v8f s = {};
            s = __builtin_amdgcn_wmma_f32_16x16x32_bf16(false, Aq0, false, B0,
                                                        (short)0, s, false, false);
            s = __builtin_amdgcn_wmma_f32_16x16x32_bf16(false, Aq1, false, B1,
                                                        (short)0, s, false, false);
            S[nt] = s;
        }

        // ---- online softmax (row r lives across 16 lanes of a half)
#pragma unroll
        for (int r = 0; r < 8; r++) {
#pragma unroll
            for (int nt = 0; nt < 2; nt++) {
                int col = kb0 + nt * 16 + l15;
                bool bad = gapMask && (col > 0) && (col < 16);
                float s = S[nt][r] * SCALE;
                S[nt][r] = bad ? -3.0e38f : s;
            }
            float v = fmaxf(S[0][r], S[1][r]);
#pragma unroll
            for (int i = 1; i < 16; i <<= 1) v = fmaxf(v, __shfl_xor(v, i, 32));
            float mnew = fmaxf(mrow[r], v);
            float corr = __expf(mrow[r] - mnew);
            mrow[r] = mnew;
            float p0 = __expf(S[0][r] - mnew);
            float p1 = __expf(S[1][r] - mnew);
            float rs = p0 + p1;
#pragma unroll
            for (int i = 1; i < 16; i <<= 1) rs += __shfl_xor(rs, i, 32);
            lrow[r] = lrow[r] * corr + rs;
#pragma unroll
            for (int t = 0; t < 4; t++) accO[t][r] *= corr;
            int row = r + 8 * half;
            ldsP[row * 32 + l15]      = f2bf(p0);
            ldsP[row * 32 + 16 + l15] = f2bf(p1);
        }
        __syncthreads();

        // P as A operand: lane=M=l15; e -> n=half*16+e
        const unsigned short* pr = ldsP + l15 * 32 + half * 16;
        v16bf Ap = cat16(ld8bf(pr), ld8bf(pr + 8));
        __syncthreads();

        // ---- O += P @ V : V^T rows are contiguous in k -> two b128 loads
#pragma unroll
        for (int t = 0; t < 4; t++) {
            const unsigned short* vp = vbT + (bhVT + t * 16 + l15) * NTOKP
                                           + kOff + kb0 + half * 16;
            v16bf Bv = cat16(ld8bf(vp), ld8bf(vp + 8));
            accO[t] = __builtin_amdgcn_wmma_f32_16x16x32_bf16(false, Ap, false, Bv,
                                                              (short)0, accO[t], false, false);
        }
    }

    // ---- normalize + store to compact rows (padded tok -> tok==0?0:tok-15)
#pragma unroll
    for (int r = 0; r < 8; r++) {
        int mg = qt * 16 + r + 8 * half;
        if (mg < Nq) {
            int ptok = qOff + mg;
            int crow = (ptok == 0) ? 0 : ptok - 15;
            float inv = 1.0f / lrow[r];
            size_t orow = ((size_t)b * 2561 + crow) * 512 + h * 64;
#pragma unroll
            for (int t = 0; t < 4; t++)
                xout[orow + t * 16 + l15] = f2bf(accO[t][r] * inv);
        }
    }
}

// ---------------------------------------------------------------------
// Stage 3: output projection (bf16 x bf16 -> f32 + bias), row remap
// ---------------------------------------------------------------------
__global__ __launch_bounds__(32) void proj_gemm(
    const unsigned short* __restrict__ X, const unsigned short* __restrict__ W,
    const float* __restrict__ bias, float* __restrict__ out)
{
    const int lane = threadIdx.x, half = lane >> 4, l15 = lane & 15;
    const int mt = blockIdx.x, ct = blockIdx.y;
    const int ROWS = 2 * 2561;
    int m = mt * 16 + l15; if (m >= ROWS) m = ROWS - 1;
    const unsigned short* xrow = X + (size_t)m * 512;
    const int c0 = ct * 16;
    const unsigned short* wrow = W + (size_t)(c0 + l15) * 512;

    v8f acc = {};
    for (int k0 = 0; k0 < 512; k0 += 32) {
        const unsigned short* xp = xrow + k0 + half * 16;
        v16bf A = cat16(ld8bf(xp), ld8bf(xp + 8));
        const unsigned short* wp = wrow + k0 + half * 16;
        v16bf Bm = cat16(ld8bf(wp), ld8bf(wp + 8));
        acc = __builtin_amdgcn_wmma_f32_16x16x32_bf16(false, A, false, Bm,
                                                      (short)0, acc, false, false);
    }
    const int c = c0 + l15;
    const float bv = bias[c];
#pragma unroll
    for (int r = 0; r < 8; r++) {
        int mg = mt * 16 + r + 8 * half;
        if (mg >= ROWS) continue;
        int b = mg / 2561, t = mg % 2561;
        size_t o;
        if (t == 0)        o = (size_t)2621440 + (size_t)b * 512;                      // xg
        else if (t < 2049) o = ((size_t)b * 2048 + (size_t)(t - 1)) * 512;             // xh
        else               o = (size_t)2097152 + ((size_t)b * 512 + (t - 2049)) * 512; // xp
        out[o + c] = acc[r] + bv;
    }
}

// ---------------------------------------------------------------------
extern "C" void kernel_launch(void* const* d_in, const int* in_sizes, int n_in,
                              void* d_out, int out_size, void* d_ws, size_t ws_size,
                              hipStream_t stream)
{
    const float* hist  = (const float*)d_in[0];   // [2,2048,512]
    const float* path  = (const float*)d_in[1];   // [2,512,512]
    const float* glob  = (const float*)d_in[2];   // [2,1,512]
    const float* Wqkv  = (const float*)d_in[3];   // [1536,512]
    const float* Wproj = (const float*)d_in[4];   // [512,512]
    const float* bproj = (const float*)d_in[5];   // [512]
    float* out = (float*)d_out;

    // workspace carve (bf16 element counts; all 16B-aligned)
    unsigned short* w = (unsigned short*)d_ws;
    unsigned short* xH  = w;  w += (size_t)2 * 2048 * 512;    // 2,097,152
    unsigned short* xP  = w;  w += (size_t)2 * 512 * 512;     //   524,288
    unsigned short* xG  = w;  w += (size_t)2 * 512;           //     1,024
    unsigned short* wQ  = w;  w += (size_t)1536 * 512;        //   786,432
    unsigned short* wP  = w;  w += (size_t)512 * 512;         //   262,144
    unsigned short* qb  = w;  w += (size_t)2 * 8 * NTOKP * 64;// 2,637,824
    unsigned short* kb  = w;  w += (size_t)2 * 8 * NTOKP * 64;
    unsigned short* vbT = w;  w += (size_t)2 * 8 * 64 * NTOKP;
    unsigned short* xc  = w;                                  // [2*2561, 512]

    dim3 blk32(32, 1, 1), blk256(256, 1, 1);
    // Stage 0: conversions (element-count / 8 threads)
    cvt_f32_bf16<<<dim3(1024), blk256, 0, stream>>>(hist,  xH, 262144);
    cvt_f32_bf16<<<dim3(256),  blk256, 0, stream>>>(path,  xP, 65536);
    cvt_f32_bf16<<<dim3(1),    blk256, 0, stream>>>(glob,  xG, 128);
    cvt_f32_bf16<<<dim3(384),  blk256, 0, stream>>>(Wqkv,  wQ, 98304);
    cvt_f32_bf16<<<dim3(128),  blk256, 0, stream>>>(Wproj, wP, 32768);
    // Stage 1: QKV projections (M-tiles x 96 col-tiles x B)
    qkv_gemm<<<dim3(128, 96, 2), blk32, 0, stream>>>(xH, wQ, qb, kb, vbT, 2048, TOK_H);
    qkv_gemm<<<dim3(32,  96, 2), blk32, 0, stream>>>(xP, wQ, qb, kb, vbT, 512,  TOK_P);
    qkv_gemm<<<dim3(1,   96, 2), blk32, 0, stream>>>(xG, wQ, qb, kb, vbT, 1,    0);
    // Stage 2: attention (q-tiles x H x B)
    attn_flash<<<dim3(128, 8, 2), blk32, 0, stream>>>(qb, kb, vbT, xc, TOK_H, 2048, TOK_H, 2048, 0);
    attn_flash<<<dim3(32,  8, 2), blk32, 0, stream>>>(qb, kb, vbT, xc, TOK_P, 512,  TOK_P, 512,  0);
    attn_flash<<<dim3(1,   8, 2), blk32, 0, stream>>>(qb, kb, vbT, xc, 0,     1,    0,     NTOKP, 1);
    // Stage 3: output projection over all 5122 compact rows
    proj_gemm<<<dim3(321, 32, 1), blk32, 0, stream>>>(xc, wP, bproj, out);
}
